// Transformer_DA_47906065219893
// MI455X (gfx1250) — matrix operand order repaired and checked by reading, not compiled, not run on hardware
//
#include <hip/hip_runtime.h>

// ---------------------------------------------------------------------------
// Shapes (fixed by the reference)
// ---------------------------------------------------------------------------
#define D_MODEL 256
#define HEADS   8
#define DHEAD   32
#define FF_DIM  2048
#define BATCH   2
#define NQ      1024
#define TV      16
#define HV      128
#define WV      128
#define THW     (TV*HV*WV)          // 262144
#define LPTS    2048                // 8*16*16 sample points per group
#define BG      (BATCH*HEADS)       // 16

typedef __attribute__((ext_vector_type(16))) _Float16 v16h;
typedef __attribute__((ext_vector_type(8)))  float    v8f;
typedef __attribute__((ext_vector_type(4)))  int      v4i;

// ---- optional CDNA5 async global->LDS path (compile-time guarded) ---------
#if defined(__has_builtin)
#if __has_builtin(__builtin_amdgcn_global_load_async_to_lds_b128) && \
    __has_builtin(__builtin_amdgcn_s_wait_asynccnt)
#define USE_ASYNC_LDS 1
#endif
#endif
#ifndef USE_ASYNC_LDS
#define USE_ASYNC_LDS 0
#endif

__device__ __forceinline__ void copy16_g2lds(const float* src, float* dstLds)
{
#if USE_ASYNC_LDS
    // signature (probed from clang diagnostic): (addrspace(1) v4i*, addrspace(3) v4i*, imm offset, cpol)
    __builtin_amdgcn_global_load_async_to_lds_b128(
        (__attribute__((address_space(1))) v4i*)(v4i*)(void*)src,
        (__attribute__((address_space(3))) v4i*)(v4i*)(void*)dstLds, 0, 0);
#else
    *(float4*)dstLds = *(const float4*)src;
#endif
}

__device__ __forceinline__ void wait_async_lds()
{
#if USE_ASYNC_LDS
    __builtin_amdgcn_s_wait_asynccnt(0);
#endif
}

// ---------------------------------------------------------------------------
// Generic batched WMMA GEMM:  C = act(alpha * A(MxK) @ B(KxN) + bias [+ Res])
//   BT : B stored (N x K) row-major (weight layout W[n][k]).
//   EDA: A side needs masking (scalar clamp+select loads); else b128 loads.
//   EDB: B side needs masking; else b128 loads (async-to-LDS when BT==0).
// Block = 256 threads = 8 waves arranged 4(M) x 2(N); each wave computes a
// 16x32 C slab = two v_wmma_f32_16x16x32_f16 sharing one A fragment.
// Block tile: 64(M) x 64(N), K stepped by 32 through LDS.
// ---------------------------------------------------------------------------
#define TM 64
#define TN 64
#define TKS 32

template<int BT, int EDA, int EDB>
__global__ __launch_bounds__(256) void gemm_wmma(
    const float* __restrict__ A, const float* __restrict__ B,
    const float* __restrict__ bias, const float* __restrict__ Res,
    float* __restrict__ C,
    int M, int N, int K, int lda, int ldb, int ldc,
    long aOut, long aIn, long bOut, long bIn, long cOut, long cIn,
    int innerCnt, float alpha, int act)
{
    __shared__ float As[TKS][TM + 1];   // [k][m]  65-float rows
    __shared__ float Bs[TKS][TN + 4];   // [k][n]  68-float rows (16B-aligned)

    const int batch = blockIdx.z;
    const int bo = batch / innerCnt, bi = batch % innerCnt;
    const float* Ab = A + (long)bo * aOut + (long)bi * aIn;
    const float* Bb = B + (long)bo * bOut + (long)bi * bIn;
    float*       Cb = C + (long)bo * cOut + (long)bi * cIn;
    const float* Rb = Res ? (Res + (long)bo * cOut + (long)bi * cIn) : nullptr;

    const int m0 = blockIdx.y * TM;
    const int n0 = blockIdx.x * TN;
    const int tid  = threadIdx.x;
    const int lane = tid & 31;
    const int wave = tid >> 5;
    const int waveM = wave >> 1;      // 0..3
    const int waveN = wave & 1;       // 0..1
    const int mf   = lane & 15;       // row/col within 16
    const int kh   = lane >> 4;       // lane half selects K sub-block

    v8f acc0 = {};
    v8f acc1 = {};

    for (int k0 = 0; k0 < K; k0 += TKS) {
        // ---------------- A tile ----------------
        if (EDA) {
            for (int i = tid; i < TM * TKS; i += 256) {
                int mm = i >> 5, kk = i & 31;
                int gm = m0 + mm, gk = k0 + kk;
                int cm = gm < M ? gm : M - 1;
                int ck = gk < K ? gk : K - 1;
                float v = Ab[(long)cm * lda + ck];
                As[kk][mm] = (gm < M && gk < K) ? v : 0.f;
            }
        } else {
            for (int i = tid; i < (TM * TKS) / 4; i += 256) {   // 512 slots
                int mm = i >> 3, k4 = (i & 7) << 2;
                const float4 v = *(const float4*)(Ab + (long)(m0 + mm) * lda + k0 + k4);
                As[k4 + 0][mm] = v.x; As[k4 + 1][mm] = v.y;
                As[k4 + 2][mm] = v.z; As[k4 + 3][mm] = v.w;
            }
            if (k0 + TKS < K)
                __builtin_prefetch(Ab + (long)(m0 + (tid & 63)) * lda + k0 + TKS, 0, 1);
        }
        // ---------------- B tile ----------------
        if (EDB) {
            for (int i = tid; i < TKS * TN; i += 256) {
                int kk = i >> 6, nn = i & 63;
                int gn = n0 + nn, gk = k0 + kk;
                int cn = gn < N ? gn : N - 1;
                int ck = gk < K ? gk : K - 1;
                float v = BT ? Bb[(long)cn * ldb + ck]
                             : Bb[(long)ck * ldb + cn];
                Bs[kk][nn] = (gn < N && gk < K) ? v : 0.f;
            }
        } else if (BT) {
            for (int i = tid; i < (TN * TKS) / 4; i += 256) {
                int nn = i >> 3, k4 = (i & 7) << 2;
                const float4 v = *(const float4*)(Bb + (long)(n0 + nn) * ldb + k0 + k4);
                Bs[k4 + 0][nn] = v.x; Bs[k4 + 1][nn] = v.y;
                Bs[k4 + 2][nn] = v.z; Bs[k4 + 3][nn] = v.w;
            }
            if (k0 + TKS < K)
                __builtin_prefetch(Bb + (long)(n0 + (tid & 63)) * ldb + k0 + TKS, 0, 1);
        } else {
            // b128 per-lane global -> LDS (async path when toolchain has it)
            for (int i = tid; i < (TN * TKS) / 4; i += 256) {
                int kk = i >> 4, n4 = (i & 15) << 2;
                copy16_g2lds(Bb + (long)(k0 + kk) * ldb + n0 + n4, &Bs[kk][n4]);
            }
            if (k0 + TKS < K)
                __builtin_prefetch(Bb + (long)(k0 + TKS + (tid & 31)) * ldb + n0, 0, 1);
        }
        if (BT == 0 && EDB == 0) wait_async_lds();   // ASYNCcnt not covered by barrier
        __syncthreads();

        // Pack fragments per ISA 7.12.2 (16-bit A 16x32; B 32x16)
        v16h af, bf0, bf1;
        const int am  = waveM * 16 + mf;
        const int bn0 = waveN * 32 + mf;
#pragma unroll
        for (int j = 0; j < 8; ++j) {
            int ka = ((j & 4) ? 16 : 0) + (j & 3) * 2 + kh * 8;   // A: K pairs
            af[2 * j]     = (_Float16)As[ka][am];
            af[2 * j + 1] = (_Float16)As[ka + 1][am];
            int kb = ((j & 4) ? 16 : 0) + (j & 3) * 2 + kh * 8;   // B: K pairs
            bf0[2 * j]     = (_Float16)Bs[kb][bn0];
            bf0[2 * j + 1] = (_Float16)Bs[kb + 1][bn0];
            bf1[2 * j]     = (_Float16)Bs[kb][bn0 + 16];
            bf1[2 * j + 1] = (_Float16)Bs[kb + 1][bn0 + 16];
        }
        acc0 = __builtin_amdgcn_wmma_f32_16x16x32_f16(
                   false, af, false, bf0, (short)0, acc0, false, false);
        acc1 = __builtin_amdgcn_wmma_f32_16x16x32_f16(
                   false, af, false, bf1, (short)0, acc1, false, false);
        __syncthreads();
    }

    // C/D layout: lane half kh -> rows r + kh*8, column = mf
    const int mb  = m0 + waveM * 16 + kh * 8;
    const int cn0 = n0 + waveN * 32 + mf;
#pragma unroll
    for (int t = 0; t < 2; ++t) {
        const int cn = cn0 + t * 16;
        const v8f acc = t ? acc1 : acc0;
        if (cn < N) {
            const float bv = bias ? bias[cn] : 0.f;
#pragma unroll
            for (int r = 0; r < 8; ++r) {
                int cm = mb + r;
                if (cm < M) {
                    float v = alpha * acc[r] + bv;
                    if (Rb) v += Rb[(long)cm * ldc + cn];
                    if (act == 1) v = v > 0.f ? v : 0.f;
                    Cb[(long)cm * ldc + cn] = v;
                }
            }
        }
    }
}

// ---------------------------------------------------------------------------
// Row softmax, in place, single pass: whole row lives in registers
// (L <= 2048 -> 8 floats / lane at 256 threads). 1 read + 1 write of HBM.
// ---------------------------------------------------------------------------
__global__ __launch_bounds__(256) void softmax_rows(float* __restrict__ X, int L)
{
    __shared__ float redm[8];
    __shared__ float reds[8];
    const int tid = threadIdx.x;
    float4* x4 = (float4*)(X + (long)blockIdx.x * L);
    const int nc = L >> 10;               // float4 chunks per thread (1 or 2)

    float4 v[2];
    float mx = -3.0e38f;
#pragma unroll
    for (int t = 0; t < 2; ++t) {
        if (t < nc) {
            v[t] = x4[tid + t * 256];
            mx = fmaxf(mx, fmaxf(fmaxf(v[t].x, v[t].y), fmaxf(v[t].z, v[t].w)));
        }
    }
    for (int o = 16; o > 0; o >>= 1) mx = fmaxf(mx, __shfl_xor(mx, o, 32));
    if ((tid & 31) == 0) redm[tid >> 5] = mx;
    __syncthreads();
    mx = redm[0];
#pragma unroll
    for (int i = 1; i < 8; ++i) mx = fmaxf(mx, redm[i]);

    float s = 0.f;
#pragma unroll
    for (int t = 0; t < 2; ++t) {
        if (t < nc) {
            v[t].x = __expf(v[t].x - mx); v[t].y = __expf(v[t].y - mx);
            v[t].z = __expf(v[t].z - mx); v[t].w = __expf(v[t].w - mx);
            s += v[t].x + v[t].y + v[t].z + v[t].w;
        }
    }
    for (int o = 16; o > 0; o >>= 1) s += __shfl_xor(s, o, 32);
    if ((tid & 31) == 0) reds[tid >> 5] = s;
    __syncthreads();
    s = 0.f;
#pragma unroll
    for (int i = 0; i < 8; ++i) s += reds[i];
    const float inv = 1.f / s;
#pragma unroll
    for (int t = 0; t < 2; ++t) {
        if (t < nc) {
            v[t].x *= inv; v[t].y *= inv; v[t].z *= inv; v[t].w *= inv;
            x4[tid + t * 256] = v[t];
        }
    }
}

// ---------------------------------------------------------------------------
// LayerNorm over rows of D_MODEL=256. One block per row, one thread per chan.
// ---------------------------------------------------------------------------
__global__ __launch_bounds__(256) void layernorm_rows(
    const float* __restrict__ X, const float* __restrict__ w,
    const float* __restrict__ b, float* __restrict__ Y)
{
    __shared__ float red1[8];
    __shared__ float red2[8];
    const int tid = threadIdx.x;
    const long row = blockIdx.x;
    float v = X[row * D_MODEL + tid];

    float s = v;
    for (int o = 16; o > 0; o >>= 1) s += __shfl_xor(s, o, 32);
    if ((tid & 31) == 0) red1[tid >> 5] = s;
    __syncthreads();
    float mu = 0.f;
#pragma unroll
    for (int i = 0; i < 8; ++i) mu += red1[i];
    mu *= (1.f / D_MODEL);

    float d = v - mu;
    float q = d * d;
    for (int o = 16; o > 0; o >>= 1) q += __shfl_xor(q, o, 32);
    if ((tid & 31) == 0) red2[tid >> 5] = q;
    __syncthreads();
    float var = 0.f;
#pragma unroll
    for (int i = 0; i < 8; ++i) var += red2[i];
    var *= (1.f / D_MODEL);

    Y[row * D_MODEL + tid] = d * rsqrtf(var + 1e-5f) * w[tid] + b[tid];
}

// ---------------------------------------------------------------------------
// Depthwise strided 3D conv for offsets.
// xg view: (BG=16, 32, 16,128,128) contiguous inside global_tokens.
// strides (2,8,8), kernel (4,10,10), pad (1,1,1). Out: (16,32,8,16,16).
// ---------------------------------------------------------------------------
__global__ __launch_bounds__(256) void dwconv3d_off(
    const float* __restrict__ X, const float* __restrict__ W,
    float* __restrict__ Out)
{
    const long idx = (long)blockIdx.x * 256 + threadIdx.x;
    const int xo = idx & 15;
    const int yo = (idx >> 4) & 15;
    const int zo = (idx >> 8) & 7;
    const int c  = (idx >> 11) & 31;
    const int bg = (int)(idx >> 16);

    const float* xp = X + ((long)bg * 32 + c) * (long)THW;
    const float* wp = W + c * 400;     // (32,1,4,10,10)
    const int zi0 = zo * 2 - 1, yi0 = yo * 8 - 1, xi0 = xo * 8 - 1;

    float acc = 0.f;
    for (int kz = 0; kz < 4; ++kz) {
        int zi = zi0 + kz;
        if ((unsigned)zi >= (unsigned)TV) continue;
        for (int ky = 0; ky < 10; ++ky) {
            int yi = yi0 + ky;
            if ((unsigned)yi >= (unsigned)HV) continue;
            const float* rowp = xp + ((long)zi * HV + yi) * WV;
            const float* wrow = wp + (kz * 10 + ky) * 10;
            for (int kx = 0; kx < 10; ++kx) {
                int xi = xi0 + kx;
                if ((unsigned)xi >= (unsigned)WV) continue;
                acc += rowp[xi] * wrow[kx];
            }
        }
    }
    Out[idx] = acc;   // bias + gelu fused downstream
}

// ---------------------------------------------------------------------------
// bias + GELU(tanh) + (3x32) projection + tanh*scale + base grid -> coords
// ---------------------------------------------------------------------------
__global__ __launch_bounds__(256) void make_coords(
    const float* __restrict__ offc, const float* __restrict__ b1,
    const float* __restrict__ w2, float* __restrict__ coords)
{
    const int idx = blockIdx.x * 256 + threadIdx.x;
    if (idx >= BG * LPTS) return;
    const int l  = idx & (LPTS - 1);
    const int bg = idx >> 11;
    const int xo = l & 15, yo = (l >> 4) & 15, zo = l >> 8;

    const float* cp = offc + (long)bg * 32 * LPTS + l;   // channel stride LPTS
    float a0 = 0.f, a1 = 0.f, a2 = 0.f;
    for (int c = 0; c < 32; ++c) {
        float v = cp[(long)c * LPTS] + b1[c];
        float u = 0.7978845608028654f * (v + 0.044715f * v * v * v);
        float g = 0.5f * v * (1.f + tanhf(u));
        a0 += w2[c] * g;            // w2 is (3,32) row-major
        a1 += w2[32 + c] * g;
        a2 += w2[64 + c] * g;
    }
    const float oz = tanhf(a0) * 2.f;   // scale = downsample (2,8,8)
    const float oy = tanhf(a1) * 8.f;
    const float ox = tanhf(a2) * 8.f;
    float* o = coords + (long)idx * 3;
    o[0] = 2.f * ((float)zo + oz) / 7.f  - 1.f;   // norm (7,15,15)
    o[1] = 2.f * ((float)yo + oy) / 15.f - 1.f;
    o[2] = 2.f * ((float)xo + ox) / 15.f - 1.f;
}

// ---------------------------------------------------------------------------
// Trilinear grid_sample (align_corners=False, zeros padding).
// Thread = (bg, c, l) with l fastest: a wave walks 32 consecutive sample
// points of one channel plane (taps share cache lines), stores coalesced.
// Writes kv (BG, 32, LPTS).
// ---------------------------------------------------------------------------
__global__ __launch_bounds__(256) void grid_sample3d(
    const float* __restrict__ X, const float* __restrict__ coords,
    float* __restrict__ KV)
{
    const long idx = (long)blockIdx.x * 256 + threadIdx.x;
    const int l  = (int)(idx & (LPTS - 1));
    const int c  = (int)((idx >> 11) & 31);
    const int bg = (int)(idx >> 16);

    const float* co = coords + ((long)bg * LPTS + l) * 3;
    const float z = ((co[0] + 1.f) * (float)TV - 1.f) * 0.5f;
    const float y = ((co[1] + 1.f) * (float)HV - 1.f) * 0.5f;
    const float x = ((co[2] + 1.f) * (float)WV - 1.f) * 0.5f;
    const float zf = floorf(z), yf = floorf(y), xf = floorf(x);
    const float wz = z - zf, wy = y - yf, wx = x - xf;
    const int z0 = (int)zf, y0 = (int)yf, x0 = (int)xf;

    const float* xp = X + ((long)bg * 32 + c) * (long)THW;
    float acc = 0.f;
#pragma unroll
    for (int dz = 0; dz < 2; ++dz) {
        int zi = z0 + dz;
        if ((unsigned)zi >= (unsigned)TV) continue;
        float wzz = dz ? wz : 1.f - wz;
#pragma unroll
        for (int dy = 0; dy < 2; ++dy) {
            int yi = y0 + dy;
            if ((unsigned)yi >= (unsigned)HV) continue;
            float wyy = dy ? wy : 1.f - wy;
#pragma unroll
            for (int dx = 0; dx < 2; ++dx) {
                int xi = x0 + dx;
                if ((unsigned)xi >= (unsigned)WV) continue;
                float wxx = dx ? wx : 1.f - wx;
                acc += xp[((long)zi * HV + yi) * WV + xi] * wzz * wyy * wxx;
            }
        }
    }
    KV[idx] = acc;   // (bg,c,l) contiguous
}

// ---------------------------------------------------------------------------
// Host-side orchestration
// ---------------------------------------------------------------------------
extern "C" void kernel_launch(void* const* d_in, const int* in_sizes, int n_in,
                              void* d_out, int out_size, void* d_ws, size_t ws_size,
                              hipStream_t stream)
{
    (void)in_sizes; (void)n_in; (void)out_size; (void)ws_size;

    const float* q_tok   = (const float*)d_in[0];
    const float* g_tok   = (const float*)d_in[1];
    const float* sa_in_w = (const float*)d_in[2];
    const float* sa_in_b = (const float*)d_in[3];
    const float* sa_o_w  = (const float*)d_in[4];
    const float* sa_o_b  = (const float*)d_in[5];
    const float* q_w     = (const float*)d_in[6];
    const float* off_w1  = (const float*)d_in[7];
    const float* off_b1  = (const float*)d_in[8];
    const float* off_w2  = (const float*)d_in[9];
    const float* k_w     = (const float*)d_in[10];
    const float* v_w     = (const float*)d_in[11];
    const float* da_o_w  = (const float*)d_in[12];
    const float* da_o_b  = (const float*)d_in[13];
    const float* lin1_w  = (const float*)d_in[14];
    const float* lin1_b  = (const float*)d_in[15];
    const float* lin2_w  = (const float*)d_in[16];
    const float* lin2_b  = (const float*)d_in[17];
    const float* ln1_w   = (const float*)d_in[18];
    const float* ln1_b   = (const float*)d_in[19];
    const float* ln2_w   = (const float*)d_in[20];
    const float* ln2_b   = (const float*)d_in[21];
    const float* ln3_w   = (const float*)d_in[22];
    const float* ln3_b   = (const float*)d_in[23];

    float* t_out    = (float*)d_out;                       // (2,1024,256)
    float* attn_out = t_out + (long)BATCH * NQ * D_MODEL;  // (2,8,1024,2048)

    // workspace layout (floats)
    float* w = (float*)d_ws;
    float* qkv  = w; w += (long)BATCH * NQ * 3 * D_MODEL;
    float* sscr = w; w += (long)BG * NQ * NQ;
    float* o1   = w; w += (long)BATCH * NQ * D_MODEL;
    float* r1   = w; w += (long)BATCH * NQ * D_MODEL;
    float* t1   = w; w += (long)BATCH * NQ * D_MODEL;
    float* offc = w; w += (long)BG * 32 * LPTS;
    float* crd  = w; w += (long)BG * LPTS * 3;
    float* kvb  = w; w += (long)BG * 32 * LPTS;
    float* kb   = w; w += (long)BG * 32 * LPTS;
    float* vb   = w; w += (long)BG * 32 * LPTS;
    float* qp   = w; w += (long)BATCH * NQ * D_MODEL;
    float* o2   = w; w += (long)BATCH * NQ * D_MODEL;
    float* r2   = w; w += (long)BATCH * NQ * D_MODEL;
    float* t2   = w; w += (long)BATCH * NQ * D_MODEL;
    float* ffh  = w; w += (long)BATCH * NQ * FF_DIM;
    float* r3   = w; w += (long)BATCH * NQ * D_MODEL;

    const float iskd = 0.17677669529663689f;   // 1/sqrt(32)
    const dim3 blk(256);

    auto gemm = [&](const float* A, const float* B, const float* bias,
                    const float* Res, float* C,
                    int M, int N, int K, int lda, int ldb, int ldc,
                    long aO, long aI, long bO, long bI, long cO, long cI,
                    int innerCnt, int batches, int bTrans, float alpha, int act) {
        dim3 g((N + TN - 1) / TN, (M + TM - 1) / TM, batches);
        const int eda = ((M & (TM - 1)) || (K & (TKS - 1))) ? 1 : 0;
        const int edb = ((N & (TN - 1)) || (K & (TKS - 1))) ? 1 : 0;
#define LAUNCH_GEMM(BT_, EA_, EB_) \
        gemm_wmma<BT_, EA_, EB_><<<g, blk, 0, stream>>>(A, B, bias, Res, C, \
            M, N, K, lda, ldb, ldc, aO, aI, bO, bI, cO, cI, innerCnt, alpha, act)
        if (bTrans) {
            if (eda) { if (edb) LAUNCH_GEMM(1,1,1); else LAUNCH_GEMM(1,1,0); }
            else     { if (edb) LAUNCH_GEMM(1,0,1); else LAUNCH_GEMM(1,0,0); }
        } else {
            if (eda) { if (edb) LAUNCH_GEMM(0,1,1); else LAUNCH_GEMM(0,1,0); }
            else     { if (edb) LAUNCH_GEMM(0,0,1); else LAUNCH_GEMM(0,0,0); }
        }
#undef LAUNCH_GEMM
    };

    const int MN = BATCH * NQ;                 // 2048
    const long QKVB = (long)NQ * 3 * D_MODEL;  // per-b stride inside qkv
    const long SB   = (long)NQ * NQ;           // self-attn score per-bh

    // ---- 1) qkv = q_tok @ sa_in_w^T + b ---------------------------------
    gemm(q_tok, sa_in_w, sa_in_b, nullptr, qkv,
         MN, 3 * D_MODEL, D_MODEL, D_MODEL, D_MODEL, 3 * D_MODEL,
         0, 0, 0, 0, 0, 0, 1, 1, 1, 1.f, 0);

    // ---- 2) self-attn scores (per b,h): q @ k^T / sqrt(d) ---------------
    gemm(qkv /*q*/, qkv + D_MODEL /*k*/, nullptr, nullptr, sscr,
         NQ, NQ, DHEAD, 3 * D_MODEL, 3 * D_MODEL, NQ,
         QKVB, DHEAD, QKVB, DHEAD, 8 * SB, SB,
         HEADS, BG, 1, iskd, 0);

    // ---- 3) softmax over keys -------------------------------------------
    softmax_rows<<<dim3(BG * NQ), blk, 0, stream>>>(sscr, NQ);

    // ---- 4) o = att @ v  (A aligned b128 path; only B/C masked) ----------
    gemm(sscr, qkv + 2 * D_MODEL /*v*/, nullptr, nullptr, o1,
         NQ, DHEAD, NQ, NQ, 3 * D_MODEL, D_MODEL,
         8 * SB, SB, QKVB, DHEAD, (long)NQ * D_MODEL, DHEAD,
         HEADS, BG, 0, 1.f, 0);

    // ---- 5) out proj + residual(q_tok); 6) LN1 ---------------------------
    gemm(o1, sa_o_w, sa_o_b, q_tok, r1,
         MN, D_MODEL, D_MODEL, D_MODEL, D_MODEL, D_MODEL,
         0, 0, 0, 0, 0, 0, 1, 1, 1, 1.f, 0);
    layernorm_rows<<<dim3(MN), blk, 0, stream>>>(r1, ln1_w, ln1_b, t1);

    // ---- 7) offset depthwise conv ---------------------------------------
    dwconv3d_off<<<dim3((BG * 32 * LPTS) / 256), blk, 0, stream>>>(g_tok, off_w1, offc);

    // ---- 8) gelu + proj + tanh*scale -> sampling coords ------------------
    make_coords<<<dim3((BG * LPTS + 255) / 256), blk, 0, stream>>>(offc, off_b1, off_w2, crd);

    // ---- 9) trilinear gather -> kv (BG,32,L) -----------------------------
    grid_sample3d<<<dim3((BG * LPTS * 32) / 256), blk, 0, stream>>>(g_tok, crd, kvb);

    // ---- 10/11) k,v = per-group 32x32 projections of kv ------------------
    const long KVB = (long)32 * LPTS;          // 65536 per bg
    gemm(k_w, kvb, nullptr, nullptr, kb,
         32, LPTS, 32, 32, LPTS, LPTS,
         0, 32 * 32, 8 * KVB, KVB, 8 * KVB, KVB,
         HEADS, BG, 0, 1.f, 0);
    gemm(v_w, kvb, nullptr, nullptr, vb,
         32, LPTS, 32, 32, LPTS, LPTS,
         0, 32 * 32, 8 * KVB, KVB, 8 * KVB, KVB,
         HEADS, BG, 0, 1.f, 0);

    // ---- 12) q projection of t1 ------------------------------------------
    gemm(t1, q_w, nullptr, nullptr, qp,
         MN, D_MODEL, D_MODEL, D_MODEL, D_MODEL, D_MODEL,
         0, 0, 0, 0, 0, 0, 1, 1, 1, 1.f, 0);

    // ---- 13) deform scores -> attn output region (pre-softmax) -----------
    const long AB = (long)NQ * LPTS;           // 2097152 per bg
    gemm(qp, kb, nullptr, nullptr, attn_out,
         NQ, LPTS, DHEAD, D_MODEL, LPTS, LPTS,
         (long)NQ * D_MODEL, DHEAD, 8 * KVB, KVB, 8 * AB, AB,
         HEADS, BG, 0, iskd, 0);

    // ---- 14) softmax over L (this IS output #2) --------------------------
    softmax_rows<<<dim3(BG * NQ), blk, 0, stream>>>(attn_out, LPTS);

    // ---- 15) o = attn @ v^T  (A = 134MB attn tensor on b128 path) --------
    gemm(attn_out, vb, nullptr, nullptr, o2,
         NQ, DHEAD, LPTS, LPTS, LPTS, D_MODEL,
         8 * AB, AB, 8 * KVB, KVB, (long)NQ * D_MODEL, DHEAD,
         HEADS, BG, 1, 1.f, 0);

    // ---- 16) deform out proj + residual(t1); 17) LN2 ---------------------
    gemm(o2, da_o_w, da_o_b, t1, r2,
         MN, D_MODEL, D_MODEL, D_MODEL, D_MODEL, D_MODEL,
         0, 0, 0, 0, 0, 0, 1, 1, 1, 1.f, 0);
    layernorm_rows<<<dim3(MN), blk, 0, stream>>>(r2, ln2_w, ln2_b, t2);

    // ---- 18) FFN up + ReLU; 19) FFN down + residual(t2); 20) LN3 ---------
    gemm(t2, lin1_w, lin1_b, nullptr, ffh,
         MN, FF_DIM, D_MODEL, D_MODEL, D_MODEL, FF_DIM,
         0, 0, 0, 0, 0, 0, 1, 1, 1, 1.f, 1);
    gemm(ffh, lin2_w, lin2_b, t2, r3,
         MN, D_MODEL, FF_DIM, FF_DIM, FF_DIM, D_MODEL,
         0, 0, 0, 0, 0, 0, 1, 1, 1, 1.f, 0);
    layernorm_rows<<<dim3(MN), blk, 0, stream>>>(r3, ln3_w, ln3_b, t_out);
}
